// NaiveFourierKANLayer_1743756722951
// MI455X (gfx1250) — matrix operand order repaired and checked by reading, not compiled, not run on hardware
//
#include <hip/hip_runtime.h>
#include <stdint.h>

#define BATCH 4096
#define IDIM  256
#define ODIM  256
#define GRIDG 64

#define MTILE 64
#define NTILE 128
#define APITCH 136              // halves per A row (128 + 8 pad) = 272B = 16*17
#define BPITCH 136              // halves per B row (128 + 8 pad)

typedef _Float16 v16h __attribute__((ext_vector_type(16)));
typedef _Float16 v8h  __attribute__((ext_vector_type(8)));
typedef float    v8f  __attribute__((ext_vector_type(8)));

static __device__ __forceinline__ uint32_t pk2h(float a, float b) {
  union { _Float16 h[2]; uint32_t u; } z;
  z.h[0] = (_Float16)a; z.h[1] = (_Float16)b;
  return z.u;
}

// ---------------------------------------------------------------------------
// Pre-pass: repack fc f32 [2][O][I][G] -> f16 [O][I][2*G]
// For each (o,i): 128 contiguous halves = cos coeffs g=0..63, sin coeffs g=0..63
// (exactly the LDS B-tile row layout, so the GEMM can async-DMA it directly).
// ---------------------------------------------------------------------------
__global__ __launch_bounds__(256)
void convert_coeffs(const float* __restrict__ fc, _Float16* __restrict__ w16)
{
  const size_t id = (size_t)blockIdx.x * 256 + threadIdx.x;  // 1M ids, 8 elems each
  const int j = (int)(id & 7);        // 8-float segment within g-run
  size_t r = id >> 3;
  const int p = (int)(r & 1);         // 0 = cos, 1 = sin
  r >>= 1;
  const int i = (int)(r & (IDIM - 1));
  const int o = (int)(r >> 8);

  const float* src = fc + (((size_t)p * ODIM + o) * IDIM + i) * GRIDG + j * 8;
  const float4 v0 = ((const float4*)src)[0];
  const float4 v1 = ((const float4*)src)[1];

  uint32_t* dst = (uint32_t*)w16 + id * 4;   // dst halves offset = id*8
  dst[0] = pk2h(v0.x, v0.y);
  dst[1] = pk2h(v0.z, v0.w);
  dst[2] = pk2h(v1.x, v1.y);
  dst[3] = pk2h(v1.z, v1.w);
}

// ---------------------------------------------------------------------------
// Main GEMM: B tiles DMA'd memory->LDS with GLOBAL_LOAD_ASYNC_TO_LDS_B128,
// overlapped with the sincos A-tile generation.
// ---------------------------------------------------------------------------
__global__ __launch_bounds__(256)
void fourier_kan_wmma_async(const float* __restrict__ x,
                            const _Float16* __restrict__ w16,
                            const float* __restrict__ bias,
                            float* __restrict__ out)
{
  __shared__ _Float16 Alds[MTILE * APITCH];   // [m][k]  k: 0..63 cos, 64..127 sin
  __shared__ _Float16 Blds[NTILE * BPITCH];   // [n][k]  same k layout

  const int tid  = threadIdx.x;
  const int lane = tid & 31;
  const int w    = tid >> 5;
  const int lo   = lane & 15;
  const int hi   = lane >> 4;

  const int row_base = blockIdx.x * MTILE;
  const int o_base   = blockIdx.y * NTILE;

  const int m0 = (w >> 1) * 16;
  const int n0 = (w & 1) * 64;

  // A-staging mapping: 256 threads cover 64 rows x 64 angles (each -> cos+sin)
  const int sa_m = tid >> 2;
  const int sa_q = tid & 3;

  // B async-staging mapping: 2048 16B lane-transfers (128 cols x 256B), 8/thread
  const int seg = tid & 15;           // 16B segment within a column's 256B run
  const int nb  = tid >> 4;           // base column 0..15 (+ j*16)
  const uint32_t blds_base = (uint32_t)(uintptr_t)&Blds[0];
  const uint64_t w16b = (uint64_t)(uintptr_t)w16;
  uint32_t lds_b[8];
  uint64_t g_b[8];
  #pragma unroll
  for (int j = 0; j < 8; ++j) {
    const int n = nb + j * 16;
    lds_b[j] = blds_base + (uint32_t)(n * BPITCH + seg * 8) * 2u;
    g_b[j]   = w16b + (uint64_t)(o_base + n) * IDIM * 256u + (uint64_t)seg * 16u;
  }

  v8f acc[4] = {};

  for (int i = 0; i < IDIM; ++i) {
    // ---- kick off B tile DMA: global -> LDS, no VGPR data movement ----
    #pragma unroll
    for (int j = 0; j < 8; ++j) {
      const uint64_t ga = g_b[j] + (uint64_t)i * 256u;
      asm volatile("global_load_async_to_lds_b128 %0, %1, off"
                   :: "v"(lds_b[j]), "v"(ga)
                   : "memory");
    }
    if (i + 1 < IDIM) {
      __builtin_prefetch((const void*)(uintptr_t)(g_b[0] + (uint64_t)(i + 1) * 256u), 0, 1);
    }

    // ---- stage A under the DMA: one sincos per angle fills both K halves ----
    {
      const float xv = x[(size_t)(row_base + sa_m) * IDIM + i];
      uint32_t* arow_c = (uint32_t*)&Alds[sa_m * APITCH + sa_q * 16];
      uint32_t* arow_s = (uint32_t*)&Alds[sa_m * APITCH + 64 + sa_q * 16];
      #pragma unroll
      for (int j = 0; j < 8; ++j) {
        const float f0 = xv * (float)(sa_q * 16 + 2 * j + 1);
        const float f1 = xv * (float)(sa_q * 16 + 2 * j + 2);
        float s0, c0, s1, c1;
        __sincosf(f0, &s0, &c0);
        __sincosf(f1, &s1, &c1);
        arow_c[j] = pk2h(c0, c1);
        arow_s[j] = pk2h(s0, s1);
      }
    }

    asm volatile("s_wait_asynccnt 0x0" ::: "memory");
    __syncthreads();

    // ---- compute: 4 k-subchunks of 32 (2 cos + 2 sin), 4 N-tiles each ----
    #pragma unroll
    for (int ks = 0; ks < 4; ++ks) {
      const int kb = ks * 32;
      const _Float16* ap = &Alds[(m0 + lo) * APITCH + kb + hi * 8];
      const v8h a0 = *(const v8h*)ap;
      const v8h a1 = *(const v8h*)(ap + 16);
      const v16h a = __builtin_shufflevector(a0, a1,
          0, 1, 2, 3, 4, 5, 6, 7, 8, 9, 10, 11, 12, 13, 14, 15);
      #pragma unroll
      for (int t = 0; t < 4; ++t) {
        const _Float16* bp = &Blds[(n0 + t * 16 + lo) * BPITCH + kb + hi * 8];
        const v8h b0 = *(const v8h*)bp;
        const v8h b1 = *(const v8h*)(bp + 16);
        const v16h b = __builtin_shufflevector(b0, b1,
            0, 1, 2, 3, 4, 5, 6, 7, 8, 9, 10, 11, 12, 13, 14, 15);
        acc[t] = __builtin_amdgcn_wmma_f32_16x16x32_f16(
            false, a, false, b, (short)0, acc[t], false, false);
      }
    }

    __syncthreads();
  }

  #pragma unroll
  for (int t = 0; t < 4; ++t) {
    const int col = o_base + n0 + t * 16 + lo;
    const float bv = bias[col];
    #pragma unroll
    for (int r = 0; r < 8; ++r) {
      const int row = row_base + m0 + r + 8 * hi;
      out[(size_t)row * ODIM + col] = acc[t][r] + bv;
    }
  }
}

// ---------------------------------------------------------------------------
// Fallback (round-2 kernel): used if d_ws is too small for the f16 repack.
// ---------------------------------------------------------------------------
__global__ __launch_bounds__(256)
void fourier_kan_wmma_fb(const float* __restrict__ x,
                         const float* __restrict__ fc,
                         const float* __restrict__ bias,
                         float* __restrict__ out)
{
  __shared__ _Float16 Alds[MTILE * APITCH];
  __shared__ _Float16 Blds[NTILE * BPITCH];

  const int tid  = threadIdx.x;
  const int lane = tid & 31;
  const int w    = tid >> 5;
  const int lo   = lane & 15;
  const int hi   = lane >> 4;

  const int row_base = blockIdx.x * MTILE;
  const int o_base   = blockIdx.y * NTILE;

  const int m0 = (w >> 1) * 16;
  const int n0 = (w & 1) * 64;

  const int sa_m = tid >> 2;
  const int sa_q = tid & 3;
  const int sb_n = tid >> 1;
  const int sb_p = tid & 1;

  v8f acc[4] = {};

  for (int i = 0; i < IDIM; ++i) {
    {
      const float xv = x[(size_t)(row_base + sa_m) * IDIM + i];
      uint32_t* arow_c = (uint32_t*)&Alds[sa_m * APITCH + sa_q * 16];
      uint32_t* arow_s = (uint32_t*)&Alds[sa_m * APITCH + 64 + sa_q * 16];
      #pragma unroll
      for (int j = 0; j < 8; ++j) {
        const float f0 = xv * (float)(sa_q * 16 + 2 * j + 1);
        const float f1 = xv * (float)(sa_q * 16 + 2 * j + 2);
        float s0, c0, s1, c1;
        __sincosf(f0, &s0, &c0);
        __sincosf(f1, &s1, &c1);
        arow_c[j] = pk2h(c0, c1);
        arow_s[j] = pk2h(s0, s1);
      }
    }
    {
      const float* src = fc + (((size_t)sb_p * ODIM + o_base + sb_n) * IDIM + i) * GRIDG;
      uint32_t* brow = (uint32_t*)&Blds[sb_n * BPITCH + sb_p * 64];
      #pragma unroll
      for (int j = 0; j < 16; ++j) {
        const float4 v = ((const float4*)src)[j];
        brow[2 * j]     = pk2h(v.x, v.y);
        brow[2 * j + 1] = pk2h(v.z, v.w);
      }
      if (i + 1 < IDIM) __builtin_prefetch(src + GRIDG, 0, 1);
    }

    __syncthreads();

    #pragma unroll
    for (int ks = 0; ks < 4; ++ks) {
      const int kb = ks * 32;
      const _Float16* ap = &Alds[(m0 + lo) * APITCH + kb + hi * 8];
      const v8h a0 = *(const v8h*)ap;
      const v8h a1 = *(const v8h*)(ap + 16);
      const v16h a = __builtin_shufflevector(a0, a1,
          0, 1, 2, 3, 4, 5, 6, 7, 8, 9, 10, 11, 12, 13, 14, 15);
      #pragma unroll
      for (int t = 0; t < 4; ++t) {
        const _Float16* bp = &Blds[(n0 + t * 16 + lo) * BPITCH + kb + hi * 8];
        const v8h b0 = *(const v8h*)bp;
        const v8h b1 = *(const v8h*)(bp + 16);
        const v16h b = __builtin_shufflevector(b0, b1,
            0, 1, 2, 3, 4, 5, 6, 7, 8, 9, 10, 11, 12, 13, 14, 15);
        acc[t] = __builtin_amdgcn_wmma_f32_16x16x32_f16(
            false, a, false, b, (short)0, acc[t], false, false);
      }
    }

    __syncthreads();
  }

  #pragma unroll
  for (int t = 0; t < 4; ++t) {
    const int col = o_base + n0 + t * 16 + lo;
    const float bv = bias[col];
    #pragma unroll
    for (int r = 0; r < 8; ++r) {
      const int row = row_base + m0 + r + 8 * hi;
      out[(size_t)row * ODIM + col] = acc[t][r] + bv;
    }
  }
}

extern "C" void kernel_launch(void* const* d_in, const int* in_sizes, int n_in,
                              void* d_out, int out_size, void* d_ws, size_t ws_size,
                              hipStream_t stream) {
  const float* x    = (const float*)d_in[0];   // [4096, 256] f32
  const float* fc   = (const float*)d_in[1];   // [2, 256, 256, 64] f32
  const float* bias = (const float*)d_in[2];   // [1, 256] f32
  float* out = (float*)d_out;                  // [4096, 256] f32

  const size_t W16_BYTES = (size_t)ODIM * IDIM * 128 * 2;   // 16 MB f16 repack
  dim3 grid(BATCH / MTILE, ODIM / NTILE);                   // 64 x 2

  if (d_ws != nullptr && ws_size >= W16_BYTES) {
    _Float16* w16 = (_Float16*)d_ws;
    convert_coeffs<<<(ODIM * IDIM * 2 * GRIDG / 8) / 256, 256, 0, stream>>>(fc, w16);
    fourier_kan_wmma_async<<<grid, 256, 0, stream>>>(x, w16, bias, out);
  } else {
    fourier_kan_wmma_fb<<<grid, 256, 0, stream>>>(x, fc, bias, out);
  }

  (void)in_sizes; (void)n_in; (void)out_size; (void)ws_size;
}
